// CrossAttention_80539226735262
// MI455X (gfx1250) — matrix operand order repaired and checked by reading, not compiled, not run on hardware
//
#include <hip/hip_runtime.h>

// ---------------------------------------------------------------------------
// CrossAttention (with faithful source bugs: k unused, scores use v, reshape
// bug => attention output stored flat (B,H,L,DH) IS the (B,L,E) matrix).
//
// MI455X plan: all matmuls via v_wmma_f32_16x16x32_f16 (f32 accumulate).
// Two big GEMMs (65536x1024x1024) dominate (~274 GFLOP); attention is tiny
// (LC=77). Weights converted to f16 once (live in 192MB L2). V is
// pre-swizzled into exact WMMA B-fragment memory order so attention B-frags
// are single 32B coalesced loads.
// ---------------------------------------------------------------------------

#define B_     16
#define L_     4096
#define E_     1024
#define LC_    77
#define ECRS   768
#define H_     16
#define DH_    64
#define LCP    96      // LC padded to multiple of 32 for WMMA K / 16 for N

typedef __attribute__((ext_vector_type(16))) _Float16 v16h;
typedef __attribute__((ext_vector_type(8)))  _Float16 v8h;
typedef __attribute__((ext_vector_type(8)))  float    v8f;

__device__ __forceinline__ v8f zero8() {
  v8f z = {0.f, 0.f, 0.f, 0.f, 0.f, 0.f, 0.f, 0.f};
  return z;
}

__device__ __forceinline__ v8f wmma32(v16h a, v16h b, v8f c) {
  // D = A(16x32 f16) * B(32x16 f16) + C(f32)
  return __builtin_amdgcn_wmma_f32_16x16x32_f16(false, a, false, b,
                                                (short)0, c, false, false);
}

__device__ __forceinline__ v16h cat16(v8h lo, v8h hi) {
  return __builtin_shufflevector(lo, hi, 0, 1, 2, 3, 4, 5, 6, 7,
                                 8, 9, 10, 11, 12, 13, 14, 15);
}

// A-fragment 16x32 f16 from row-major f16 (caller pre-offsets to tile origin).
// Contiguous k order used here (same as load_bw): lane L -> row (L&15); halves 0..15 = K[base..base+15],
// base = (L>=16)*16.
__device__ __forceinline__ v16h load_a16(const _Float16* A, int ldk, int lane) {
  const int m  = lane & 15;
  const int kb = (lane >> 4) << 4;
  const _Float16* p = A + (size_t)m * ldk + kb;
  return cat16(*(const v8h*)p, *(const v8h*)(p + 8));
}

// Same A-fragment but converting from f32 source on the fly (for x).
__device__ __forceinline__ v16h load_a32(const float* A, int ldk, int lane) {
  const int m  = lane & 15;
  const int kb = (lane >> 4) << 4;
  const float* p = A + (size_t)m * ldk + kb;
  v16h r;
#pragma unroll
  for (int i = 0; i < 8; ++i) r[i] = (_Float16)p[i];
#pragma unroll
  for (int i = 0; i < 8; ++i) r[i + 8] = (_Float16)p[i + 8];
  return r;
}

// B-fragment 32x16 with B[k][n] = W[n][k], W row-major (N_out x K) f16.
// Contiguous k order used here (same as load_a16): lane L -> column (L&15), halves h = K[(L>>4)*16 + h]
// => one contiguous 32B chunk of weight row n.
__device__ __forceinline__ v16h load_bw(const _Float16* W, int ldk, int lane) {
  const int n  = lane & 15;
  const int kb = (lane >> 4) << 4;
  const _Float16* p = W + (size_t)n * ldk + kb;
  return cat16(*(const v8h*)p, *(const v8h*)(p + 8));
}

// B-fragment from pre-swizzled storage: frag = [lane][16 halves].
__device__ __forceinline__ v16h load_bp(const _Float16* F, int lane) {
  const _Float16* p = F + lane * 16;
  return cat16(*(const v8h*)p, *(const v8h*)(p + 8));
}

// ---------------------------------------------------------------------------
// K1: f32 -> f16 weight conversion
// ---------------------------------------------------------------------------
__global__ void cvt_kernel(const float* __restrict__ s, _Float16* __restrict__ d,
                           int n) {
  int i = blockIdx.x * blockDim.x + threadIdx.x;
  if (i < n) d[i] = (_Float16)s[i];
}

// ---------------------------------------------------------------------------
// K2a: v projection (tiny: 1.9 GFLOP): vtmp[bh][c][d] f32
// ---------------------------------------------------------------------------
__global__ void vproj_kernel(const float* __restrict__ cross,
                             const float* __restrict__ v_w,
                             const float* __restrict__ v_b,
                             float* __restrict__ vtmp) {
  const int NT = B_ * H_ * LC_ * DH_;
  int idx = blockIdx.x * blockDim.x + threadIdx.x;
  if (idx >= NT) return;
  const int d  = idx & 63;
  const int c  = (idx >> 6) % LC_;
  const int bh = idx / (LC_ * DH_);
  const int b  = bh >> 4, h = bh & 15;
  const int f  = h * DH_ + d;
  const float4* xr = (const float4*)(cross + ((size_t)b * LC_ + c) * ECRS);
  const float4* wr = (const float4*)(v_w + (size_t)f * ECRS);
  float acc = v_b[f];
#pragma unroll 4
  for (int k = 0; k < ECRS / 4; ++k) {
    float4 xv = xr[k], wv = wr[k];
    acc += xv.x * wv.x + xv.y * wv.y + xv.z * wv.z + xv.w * wv.w;
  }
  vtmp[idx] = acc;
}

// ---------------------------------------------------------------------------
// K2b: swizzle v into WMMA B-fragment memory order (f16, zero-padded to 96).
// vA: B[k=d][n=c] frags for scores (2 ksteps x 6 ctiles per bh)
// vB: B[k=c][n=d] frags for attn*v (3 ksteps x 4 ntiles per bh)
// frag element: [lane][h]: value at k = s*32 + (lane>>4)*16 + h, n = tile*16 + (lane&15)
// ---------------------------------------------------------------------------
__global__ void vswz_kernel(const float* __restrict__ vtmp,
                            _Float16* __restrict__ vA,
                            _Float16* __restrict__ vB) {
  const int NA = B_ * H_ * 12 * 512;  // 1,572,864 halves each
  int idx = blockIdx.x * blockDim.x + threadIdx.x;
  if (idx >= 2 * NA) return;
  if (idx < NA) {
    const int hh   = idx & 15;
    const int lane = (idx >> 4) & 31;
    int rem = idx >> 9;
    const int ct = rem % 6; rem /= 6;
    const int s  = rem & 1;
    const int bh = rem >> 1;
    const int c = ct * 16 + (lane & 15);
    const int d = s * 32 + ((lane >> 4) << 4) + hh;
    const float v = (c < LC_) ? vtmp[((size_t)bh * LC_ + c) * DH_ + d] : 0.f;
    vA[idx] = (_Float16)v;
  } else {
    const int j    = idx - NA;
    const int hh   = j & 15;
    const int lane = (j >> 4) & 31;
    int rem = j >> 9;
    const int nt = rem & 3; rem >>= 2;
    const int s  = rem % 3;
    const int bh = rem / 3;
    const int d = nt * 16 + (lane & 15);
    const int c = s * 32 + ((lane >> 4) << 4) + hh;
    const float v = (c < LC_) ? vtmp[((size_t)bh * LC_ + c) * DH_ + d] : 0.f;
    vB[j] = (_Float16)v;
  }
}

// ---------------------------------------------------------------------------
// K3: Q projection GEMM. M=65536, N=1024, K=1024.
// 256 threads = 8 waves (2x4), wave tile 64x64 = 4x4 WMMA tiles, block 128x256.
// A (x) converted f32->f16 on the fly; x is 256 MB f32 and is read by all 4 column blocks.
// Output written f16 directly into attention layout (B,H,L,DH).
// ---------------------------------------------------------------------------
__global__ __launch_bounds__(256)
void qproj_kernel(const float* __restrict__ x, const _Float16* __restrict__ qwh,
                  const float* __restrict__ q_b, _Float16* __restrict__ qh) {
  const int lane = threadIdx.x & 31;
  const int wave = threadIdx.x >> 5;
  const int wm = wave >> 2, wn = wave & 3;
  const size_t m0 = (size_t)blockIdx.x * 128 + wm * 64;
  const int n0 = blockIdx.y * 256 + wn * 64;

  v8f c[4][4];
#pragma unroll
  for (int i = 0; i < 4; ++i)
#pragma unroll
    for (int j = 0; j < 4; ++j) c[i][j] = zero8();

  for (int k0 = 0; k0 < E_; k0 += 32) {
    v16h a[4], b[4];
#pragma unroll
    for (int mt = 0; mt < 4; ++mt)
      a[mt] = load_a32(x + (m0 + mt * 16) * E_ + k0, E_, lane);
#pragma unroll
    for (int nt = 0; nt < 4; ++nt)
      b[nt] = load_bw(qwh + (size_t)(n0 + nt * 16) * E_ + k0, E_, lane);
#pragma unroll
    for (int mt = 0; mt < 4; ++mt)
#pragma unroll
      for (int nt = 0; nt < 4; ++nt)
        c[mt][nt] = wmma32(a[mt], b[nt], c[mt][nt]);
  }

#pragma unroll
  for (int nt = 0; nt < 4; ++nt) {
    const int f = n0 + nt * 16 + (lane & 15);
    const float bias = q_b[f];
    const int h = f >> 6, d = f & 63;
#pragma unroll
    for (int mt = 0; mt < 4; ++mt) {
      const size_t mb = m0 + mt * 16 + ((lane >> 4) << 3);
#pragma unroll
      for (int r = 0; r < 8; ++r) {
        const size_t m = mb + r;
        const size_t bb = m >> 12, l = m & 4095;
        qh[(((bb * H_) + h) * (size_t)L_ + l) * DH_ + d] =
            (_Float16)(c[mt][nt][r] + bias);
      }
    }
  }
}

// ---------------------------------------------------------------------------
// K4: attention per (b,h): 64-row tile per block, 2 waves x 32 rows.
// scores = q * v^T / 8 (K=64, N=96 padded) -> LDS -> row softmax (LC=77)
// -> attn(f16 in LDS, A-frags via ds_load_b128) * v (K=96, N=64).
// Output stored flat (B,H,L,DH) f16 == (B,L,E) matrix (reference reshape bug).
// ---------------------------------------------------------------------------
#define AROWS 64
__global__ __launch_bounds__(64)
void attn_kernel(const _Float16* __restrict__ qh, const _Float16* __restrict__ vA,
                 const _Float16* __restrict__ vB, _Float16* __restrict__ qkvh) {
  extern __shared__ char smem[];
  float*    s_sc = (float*)smem;                          // [64][96] f32
  _Float16* s_at = (_Float16*)(smem + AROWS * LCP * 4);   // [64][96] f16

  const int lane = threadIdx.x & 31;
  const int wave = threadIdx.x >> 5;
  const int bh = blockIdx.x >> 6;
  const int lt = blockIdx.x & 63;
  const size_t rowg0 = (size_t)bh * L_ + (size_t)lt * AROWS;
  const _Float16* qbase = qh + rowg0 * DH_;
  const int wrow = wave * 32;

  // ---- phase 1: scores tiles via WMMA ----
  v16h a[2][2];
#pragma unroll
  for (int mt = 0; mt < 2; ++mt)
#pragma unroll
    for (int s = 0; s < 2; ++s)
      a[mt][s] = load_a16(qbase + (size_t)(wrow + mt * 16) * DH_ + s * 32,
                          DH_, lane);
  const _Float16* vAb = vA + (size_t)bh * (12 * 512);
#pragma unroll
  for (int ct = 0; ct < 6; ++ct) {
    v16h b0 = load_bp(vAb + (0 * 6 + ct) * 512, lane);
    v16h b1 = load_bp(vAb + (1 * 6 + ct) * 512, lane);
#pragma unroll
    for (int mt = 0; mt < 2; ++mt) {
      v8f cc = zero8();
      cc = wmma32(a[mt][0], b0, cc);
      cc = wmma32(a[mt][1], b1, cc);
      const int col = ct * 16 + (lane & 15);
      const int rb  = wrow + mt * 16 + ((lane >> 4) << 3);
#pragma unroll
      for (int r = 0; r < 8; ++r) s_sc[(rb + r) * LCP + col] = cc[r];
    }
  }
  __syncthreads();

  // ---- phase 2: softmax over LC=77 valid columns, one row per thread ----
  {
    const int row = threadIdx.x;
    const float scale = 0.125f;  // 1/sqrt(64)
    float mx = -3.0e38f;
    for (int ci = 0; ci < LC_; ++ci) mx = fmaxf(mx, s_sc[row * LCP + ci]);
    float sum = 0.f;
    for (int ci = 0; ci < LC_; ++ci) {
      const float e = __expf((s_sc[row * LCP + ci] - mx) * scale);
      s_sc[row * LCP + ci] = e;
      sum += e;
    }
    const float inv = 1.f / sum;
    for (int ci = 0; ci < LC_; ++ci)
      s_at[row * LCP + ci] = (_Float16)(s_sc[row * LCP + ci] * inv);
    for (int ci = LC_; ci < LCP; ++ci) s_at[row * LCP + ci] = (_Float16)0.f;
  }
  __syncthreads();

  // ---- phase 3: qkv = attn * v via WMMA (A-frags from LDS) ----
  v16h a2[2][3];
#pragma unroll
  for (int mt = 0; mt < 2; ++mt)
#pragma unroll
    for (int s = 0; s < 3; ++s)
      a2[mt][s] = load_a16(s_at + (wrow + mt * 16) * LCP + s * 32, LCP, lane);

  const _Float16* vBb = vB + (size_t)bh * (12 * 512);
  _Float16* qout = qkvh + rowg0 * DH_;
#pragma unroll
  for (int nt = 0; nt < 4; ++nt) {
    v8f c0 = zero8(), c1 = zero8();
#pragma unroll
    for (int s = 0; s < 3; ++s) {
      v16h bb = load_bp(vBb + (s * 4 + nt) * 512, lane);
      c0 = wmma32(a2[0][s], bb, c0);
      c1 = wmma32(a2[1][s], bb, c1);
    }
    const int dcol = nt * 16 + (lane & 15);
    const int rb0  = wrow + ((lane >> 4) << 3);
#pragma unroll
    for (int r = 0; r < 8; ++r) {
      qout[(size_t)(rb0 + r) * DH_ + dcol]      = (_Float16)c0[r];
      qout[(size_t)(rb0 + 16 + r) * DH_ + dcol] = (_Float16)c1[r];
    }
  }
}

// ---------------------------------------------------------------------------
// K5: output projection GEMM (same shape as K3, A already f16, f32 out).
// ---------------------------------------------------------------------------
__global__ __launch_bounds__(256)
void oproj_kernel(const _Float16* __restrict__ qkvh,
                  const _Float16* __restrict__ owh,
                  const float* __restrict__ o_b, float* __restrict__ out) {
  const int lane = threadIdx.x & 31;
  const int wave = threadIdx.x >> 5;
  const int wm = wave >> 2, wn = wave & 3;
  const size_t m0 = (size_t)blockIdx.x * 128 + wm * 64;
  const int n0 = blockIdx.y * 256 + wn * 64;

  v8f c[4][4];
#pragma unroll
  for (int i = 0; i < 4; ++i)
#pragma unroll
    for (int j = 0; j < 4; ++j) c[i][j] = zero8();

  for (int k0 = 0; k0 < E_; k0 += 32) {
    v16h a[4], b[4];
#pragma unroll
    for (int mt = 0; mt < 4; ++mt)
      a[mt] = load_a16(qkvh + (m0 + mt * 16) * E_ + k0, E_, lane);
#pragma unroll
    for (int nt = 0; nt < 4; ++nt)
      b[nt] = load_bw(owh + (size_t)(n0 + nt * 16) * E_ + k0, E_, lane);
#pragma unroll
    for (int mt = 0; mt < 4; ++mt)
#pragma unroll
      for (int nt = 0; nt < 4; ++nt)
        c[mt][nt] = wmma32(a[mt], b[nt], c[mt][nt]);
  }

#pragma unroll
  for (int nt = 0; nt < 4; ++nt) {
    const int n = n0 + nt * 16 + (lane & 15);
    const float bias = o_b[n];
#pragma unroll
    for (int mt = 0; mt < 4; ++mt) {
      const size_t mb = m0 + mt * 16 + ((lane >> 4) << 3);
#pragma unroll
      for (int r = 0; r < 8; ++r)
        out[(mb + r) * E_ + n] = c[mt][nt][r] + bias;
    }
  }
}

// ---------------------------------------------------------------------------
// Workspace layout (bytes): total ~284 MB
// ---------------------------------------------------------------------------
#define WS_QH    ((size_t)0)            // q f16 (B,H,L,DH):   134,217,728
#define WS_QKVH  ((size_t)134217728)    // qkv f16 flat:       134,217,728
#define WS_QWH   ((size_t)268435456)    // q_w f16:              2,097,152
#define WS_OWH   ((size_t)270532608)    // o_w f16:              2,097,152
#define WS_VTMP  ((size_t)272629760)    // v f32 (BH,77,64):     5,046,272
#define WS_VA    ((size_t)277676032)    // v frags (scores B):   3,145,728
#define WS_VB    ((size_t)280821760)    // v frags (attnV B):    3,145,728

extern "C" void kernel_launch(void* const* d_in, const int* in_sizes, int n_in,
                              void* d_out, int out_size, void* d_ws,
                              size_t ws_size, hipStream_t stream) {
  (void)in_sizes; (void)n_in; (void)out_size; (void)ws_size;
  const float* x     = (const float*)d_in[0];
  const float* cross = (const float*)d_in[1];
  const float* q_w   = (const float*)d_in[2];
  const float* q_b   = (const float*)d_in[3];
  // d_in[4] (k_w), d_in[5] (k_b): dead code in reference -> skipped
  const float* v_w   = (const float*)d_in[6];
  const float* v_b   = (const float*)d_in[7];
  const float* o_w   = (const float*)d_in[8];
  const float* o_b   = (const float*)d_in[9];
  float* out = (float*)d_out;

  char* ws = (char*)d_ws;
  _Float16* qh   = (_Float16*)(ws + WS_QH);
  _Float16* qkvh = (_Float16*)(ws + WS_QKVH);
  _Float16* qwh  = (_Float16*)(ws + WS_QWH);
  _Float16* owh  = (_Float16*)(ws + WS_OWH);
  float*    vtmp = (float*)(ws + WS_VTMP);
  _Float16* vA   = (_Float16*)(ws + WS_VA);
  _Float16* vB   = (_Float16*)(ws + WS_VB);

  // weight conversions (f32 -> f16)
  cvt_kernel<<<4096, 256, 0, stream>>>(q_w, qwh, E_ * E_);
  cvt_kernel<<<4096, 256, 0, stream>>>(o_w, owh, E_ * E_);

  // v projection + fragment swizzle
  vproj_kernel<<<(B_ * H_ * LC_ * DH_ + 255) / 256, 256, 0, stream>>>(
      cross, v_w, v_b, vtmp);
  vswz_kernel<<<(2 * B_ * H_ * 12 * 512 + 255) / 256, 256, 0, stream>>>(
      vtmp, vA, vB);

  // q projection: M=65536, N=1024 -> grid (512, 4), 256 thr
  qproj_kernel<<<dim3(512, 4, 1), 256, 0, stream>>>(x, qwh, q_b, qh);

  // attention: (B*H) * (L/64) blocks, 64 threads, 36 KB dynamic LDS
  attn_kernel<<<B_ * H_ * (L_ / AROWS), AROWS,
                AROWS * LCP * 4 + AROWS * LCP * 2, stream>>>(qh, vA, vB, qkvh);

  // output projection
  oproj_kernel<<<dim3(512, 4, 1), 256, 0, stream>>>(qkvh, owh, o_b, out);
}
